// AttentionDecoder_16106127360233
// MI455X (gfx1250) — matrix operand order repaired and checked by reading, not compiled, not run on hardware
//
#include <hip/hip_runtime.h>
#include <hip/hip_bf16.h>

// ---------------------------------------------------------------------------
// AttentionDecoder step for MI455X (gfx1250, wave32, WMMA).
// All GEMMs: v_wmma_f32_16x16x32_bf16, f32 accumulate.
// - Read-once fp32 weights (W1/W_ih/W_hh/fc_W) converted to bf16 inline.
// - W2 (reused by 2048 blocks) pre-converted to bf16 in workspace.
// - Activations (h0/embedded/context/h1) kept as bf16 so GEMM A needs no cvt.
// - K-regions are resolved host-side into flat segments so the K-loop has no
//   64-bit address multiplies (only pointer + k indexing).
// ---------------------------------------------------------------------------

typedef __attribute__((ext_vector_type(16))) __bf16 bf16x16;
typedef __attribute__((ext_vector_type(8)))  __bf16 bf16x8;
typedef __attribute__((ext_vector_type(4)))  __bf16 bf16x4;
typedef __attribute__((ext_vector_type(8)))  float  f32x8;
typedef __attribute__((ext_vector_type(4)))  float  f32x4;

#define HID   1024
#define EMBED 512
#define BATCH 64
#define SEQ   512
#define VOCAB 32000

// ---- WMMA fragment loaders (layouts per CDNA5 ISA §7.12.2) -----
// Pointer-based: caller pre-offsets by row*ld + lane K-offset; here we only
// add the compile-time element offsets so addresses stay induction variables.

// A 16x32 bf16; p already = base + row*ld + kb + ((lane&16)?8:0)
__device__ __forceinline__ bf16x16 load_a_pt(const __bf16* p) {
  bf16x8 lo = *(const bf16x8*)(p);
  bf16x8 hi = *(const bf16x8*)(p + 16);
  bf16x16 r;
#pragma unroll
  for (int i = 0; i < 8; ++i) { r[i] = lo[i]; r[8 + i] = hi[i]; }
  return r;
}

// A 16x32 bf16 from fp32 source; p already = base + row*ld + kb + ((lane&16)?8:0)
__device__ __forceinline__ bf16x16 load_a_f32_pt(const float* p) {
  f32x4 a0 = *(const f32x4*)(p);
  f32x4 a1 = *(const f32x4*)(p + 4);
  f32x4 a2 = *(const f32x4*)(p + 16);
  f32x4 a3 = *(const f32x4*)(p + 20);
  bf16x16 r;
#pragma unroll
  for (int i = 0; i < 4; ++i) {
    r[i]      = (__bf16)a0[i];
    r[4 + i]  = (__bf16)a1[i];
    r[8 + i]  = (__bf16)a2[i];
    r[12 + i] = (__bf16)a3[i];
  }
  return r;
}

// B 32x16 bf16 from fp32 W[n][k]; p already = base + n*ld + kb + ((lane&16)?8:0)
__device__ __forceinline__ bf16x16 load_b_f32_pt(const float* p) {
  f32x4 a0 = *(const f32x4*)(p);
  f32x4 a1 = *(const f32x4*)(p + 4);
  f32x4 a2 = *(const f32x4*)(p + 16);
  f32x4 a3 = *(const f32x4*)(p + 20);
  bf16x16 r;
#pragma unroll
  for (int i = 0; i < 4; ++i) {
    r[i]      = (__bf16)a0[i];
    r[4 + i]  = (__bf16)a1[i];
    r[8 + i]  = (__bf16)a2[i];
    r[12 + i] = (__bf16)a3[i];
  }
  return r;
}

// B 32x16 from bf16 source (W2); p already = base + n*ld + kb + ((lane&16)?8:0)
__device__ __forceinline__ bf16x16 load_b_pt(const __bf16* p) {
  bf16x8 lo = *(const bf16x8*)(p);
  bf16x8 hi = *(const bf16x8*)(p + 16);
  bf16x16 r;
#pragma unroll
  for (int i = 0; i < 8; ++i) { r[i] = lo[i]; r[8 + i] = hi[i]; }
  return r;
}

__device__ __forceinline__ f32x8 wmma_bf16(bf16x16 a, bf16x16 b, f32x8 c) {
  return __builtin_amdgcn_wmma_f32_16x16x32_bf16(
      false, a, false, b, (short)0, c, false, false);
}

// ---- small utility kernels -------------------------------------------------

__global__ __launch_bounds__(256) void k_cvt_bf16(const float* __restrict__ src,
                                                  __bf16* __restrict__ dst, int n4) {
  int i = blockIdx.x * blockDim.x + threadIdx.x;
  if (i < n4) {
    f32x4 x = *(const f32x4*)(src + (size_t)i * 4);
    bf16x4 o = { (__bf16)x[0], (__bf16)x[1], (__bf16)x[2], (__bf16)x[3] };
    *(bf16x4*)(dst + (size_t)i * 4) = o;
  }
}

__global__ __launch_bounds__(256) void k_zero(float* __restrict__ p, int n) {
  int i = blockIdx.x * blockDim.x + threadIdx.x;
  if (i < n) p[i] = 0.f;
}

__global__ __launch_bounds__(128) void k_embed(const int* __restrict__ tok,
                                               const float* __restrict__ emb,
                                               __bf16* __restrict__ out) {
  int b = blockIdx.x, t = threadIdx.x;
  int row = tok[b];
  f32x4 x = *(const f32x4*)(emb + (size_t)row * EMBED + t * 4);
  bf16x4 o = { (__bf16)x[0], (__bf16)x[1], (__bf16)x[2], (__bf16)x[3] };
  *(bf16x4*)(out + (size_t)b * EMBED + t * 4) = o;
}

// ---- generic M=64 WMMA GEMM: C = A * B^T (+bias0+bias1) --------------------
// Wave = 4 M-tiles x 2 N-tiles (B fragment reused 4x; weights streamed once).
// K decomposed host-side into up to 3 flat segments (A bf16, B fp32 W[n][k]).

struct Seg {
  const __bf16* A;  // region base (column 0 of the segment)
  const float*  B;  // region base (column 0 of the segment)
  int lda, ldb, len;
};

struct GemmArgs {
  Seg seg[3]; int nseg;
  const float* bias0; const float* bias1;
  float* C; int ldc; int N;
};

__global__ __launch_bounds__(256) void k_gemm(GemmArgs g) {
  const int wave = threadIdx.x >> 5;
  const int lane = threadIdx.x & 31;
  const int n0   = (blockIdx.x * 8 + wave) * 32;     // 2 N-tiles per wave
  if (n0 >= g.N) return;
  const int mrow = lane & 15;
  const int aoff = (lane & 16) ? 8 : 0;
  const int boff = (lane & 16) ? 8 : 0;

  f32x8 acc[4][2] = {};

  for (int si = 0; si < g.nseg; ++si) {
    const Seg s = g.seg[si];
    // hoist all row*ld products out of the K loop
    const __bf16* ap0 = s.A + (size_t)mrow * s.lda + aoff;
    const __bf16* ap1 = ap0 + (size_t)16 * s.lda;
    const __bf16* ap2 = ap1 + (size_t)16 * s.lda;
    const __bf16* ap3 = ap2 + (size_t)16 * s.lda;
    const float*  bp0 = s.B + (size_t)(n0 + mrow) * s.ldb + boff;
    const float*  bp1 = bp0 + (size_t)16 * s.ldb;

    for (int k = 0; k < s.len; k += 32) {
      bf16x16 a[4];
      a[0] = load_a_pt(ap0 + k);
      a[1] = load_a_pt(ap1 + k);
      a[2] = load_a_pt(ap2 + k);
      a[3] = load_a_pt(ap3 + k);
      bf16x16 b0 = load_b_f32_pt(bp0 + k);
#pragma unroll
      for (int mt = 0; mt < 4; ++mt) acc[mt][0] = wmma_bf16(a[mt], b0, acc[mt][0]);
      bf16x16 b1 = load_b_f32_pt(bp1 + k);
#pragma unroll
      for (int mt = 0; mt < 4; ++mt) acc[mt][1] = wmma_bf16(a[mt], b1, acc[mt][1]);
    }
  }

  const int mb = (lane & 16) ? 8 : 0;
#pragma unroll
  for (int t = 0; t < 2; ++t) {
    int n = n0 + t * 16 + (lane & 15);
    float bb = (g.bias0 ? g.bias0[n] : 0.f) + (g.bias1 ? g.bias1[n] : 0.f);
#pragma unroll
    for (int mt = 0; mt < 4; ++mt)
#pragma unroll
      for (int r = 0; r < 8; ++r)
        g.C[(size_t)(mt * 16 + mb + r) * g.ldc + n] = acc[mt][t][r] + bb;
  }
}

// ---- attention scores: scores[b,s] += v . tanh(dh[b] + enc[b,s] @ W2^T) ---
// Block = 64 s-rows x 256 n-cols (grid.y = 4 N-quarters); wave = 4Mx2N tiles.
// Fused tanh + dot(v) epilogue; partials merged with global atomicAdd.

__global__ __launch_bounds__(256) void k_scores(const float* __restrict__ enc,
                                                const __bf16* __restrict__ W2b,
                                                const float* __restrict__ dh,
                                                const float* __restrict__ v,
                                                float* __restrict__ scores) {
  const int wave = threadIdx.x >> 5;
  const int lane = threadIdx.x & 31;
  const int rg   = blockIdx.x;                 // 0..511 : 64-row group
  const int b    = rg >> 3;
  const int s0   = (rg & 7) << 6;
  const int nc0  = blockIdx.y * 256 + wave * 32;
  const int mrow = lane & 15;
  const int aoff = (lane & 16) ? 8 : 0;
  const int boff = (lane & 16) ? 8 : 0;

  __shared__ float red[64];
  if (threadIdx.x < 64) red[threadIdx.x] = 0.f;
  __syncthreads();

  // hoisted lane pointers
  const float* ap0 = enc + ((size_t)b * SEQ + s0 + mrow) * HID + aoff;
  const float* ap1 = ap0 + (size_t)16 * HID;
  const float* ap2 = ap1 + (size_t)16 * HID;
  const float* ap3 = ap2 + (size_t)16 * HID;
  const __bf16* bp0 = W2b + (size_t)(nc0 + mrow) * HID + boff;
  const __bf16* bp1 = bp0 + (size_t)16 * HID;

  f32x8 acc[4][2] = {};
  for (int k = 0; k < HID; k += 32) {
    bf16x16 a[4];
    a[0] = load_a_f32_pt(ap0 + k);
    a[1] = load_a_f32_pt(ap1 + k);
    a[2] = load_a_f32_pt(ap2 + k);
    a[3] = load_a_f32_pt(ap3 + k);
    bf16x16 b0 = load_b_pt(bp0 + k);
#pragma unroll
    for (int mt = 0; mt < 4; ++mt) acc[mt][0] = wmma_bf16(a[mt], b0, acc[mt][0]);
    bf16x16 b1 = load_b_pt(bp1 + k);
#pragma unroll
    for (int mt = 0; mt < 4; ++mt) acc[mt][1] = wmma_bf16(a[mt], b1, acc[mt][1]);
  }

  const float* dhb = dh + (size_t)b * HID;
  const int mb = (lane & 16) ? 8 : 0;
  float p[4][8];
#pragma unroll
  for (int mt = 0; mt < 4; ++mt)
#pragma unroll
    for (int r = 0; r < 8; ++r) p[mt][r] = 0.f;
#pragma unroll
  for (int t = 0; t < 2; ++t) {
    int n = nc0 + t * 16 + (lane & 15);
    float add = dhb[n], vn = v[n];
#pragma unroll
    for (int mt = 0; mt < 4; ++mt)
#pragma unroll
      for (int r = 0; r < 8; ++r)
        p[mt][r] += tanhf(acc[mt][t][r] + add) * vn;
  }
#pragma unroll
  for (int mt = 0; mt < 4; ++mt)
#pragma unroll
    for (int r = 0; r < 8; ++r)
      atomicAdd(&red[mt * 16 + mb + r], p[mt][r]);
  __syncthreads();
  if (threadIdx.x < 64)
    atomicAdd(&scores[(size_t)b * SEQ + s0 + threadIdx.x], red[threadIdx.x]);
}

// ---- softmax over S + context = attn @ enc (memory-bound, 134 MB) ---------

__global__ __launch_bounds__(256) void k_softmax_ctx(const float* __restrict__ scores,
                                                     const float* __restrict__ enc,
                                                     float* __restrict__ attn_out,
                                                     __bf16* __restrict__ ctx_bf) {
  const int b = blockIdx.x, tid = threadIdx.x;
  __shared__ float sc[SEQ];
  __shared__ float red[256];
  const float* srow = scores + (size_t)b * SEQ;
  float v0 = srow[tid], v1 = srow[tid + 256];
  red[tid] = fmaxf(v0, v1);
  __syncthreads();
  for (int o = 128; o > 0; o >>= 1) {
    if (tid < o) red[tid] = fmaxf(red[tid], red[tid + o]);
    __syncthreads();
  }
  float mx = red[0];
  __syncthreads();
  float e0 = __expf(v0 - mx), e1 = __expf(v1 - mx);
  red[tid] = e0 + e1;
  __syncthreads();
  for (int o = 128; o > 0; o >>= 1) {
    if (tid < o) red[tid] += red[tid + o];
    __syncthreads();
  }
  float inv = 1.f / red[0];
  float a0 = e0 * inv, a1 = e1 * inv;
  sc[tid] = a0; sc[tid + 256] = a1;
  attn_out[(size_t)b * SEQ + tid]       = a0;
  attn_out[(size_t)b * SEQ + tid + 256] = a1;
  __syncthreads();

  const float* eb = enc + (size_t)b * SEQ * HID;
  const int h = tid * 4;
  f32x4 accv = {0.f, 0.f, 0.f, 0.f};
  for (int s = 0; s < SEQ; ++s) {
    f32x4 e = *(const f32x4*)(eb + (size_t)s * HID + h);
    float w = sc[s];
    f32x4 wv = {w, w, w, w};
    accv += e * wv;
  }
  bf16x4 o = { (__bf16)accv[0], (__bf16)accv[1], (__bf16)accv[2], (__bf16)accv[3] };
  *(bf16x4*)(ctx_bf + (size_t)b * HID + h) = o;
}

// ---- LSTM pointwise update -------------------------------------------------

__device__ __forceinline__ float sigmf(float x) { return 1.f / (1.f + __expf(-x)); }

__global__ __launch_bounds__(256) void k_lstm(const float* __restrict__ gates,
                                              const float* __restrict__ c0,
                                              __bf16* __restrict__ h1_bf,
                                              float* __restrict__ out_h,
                                              float* __restrict__ out_c) {
  int id = blockIdx.x * blockDim.x + threadIdx.x;      // 0..65535
  int b = id >> 10, h = id & 1023;
  const float* gb = gates + (size_t)b * (4 * HID);
  float i = gb[h], f = gb[HID + h], g = gb[2 * HID + h], o = gb[3 * HID + h];
  float c1 = sigmf(f) * c0[id] + sigmf(i) * tanhf(g);
  float h1 = sigmf(o) * tanhf(c1);
  h1_bf[id] = (__bf16)h1;
  out_h[id] = h1;
  out_c[id] = c1;
}

// ---------------------------------------------------------------------------

extern "C" void kernel_launch(void* const* d_in, const int* in_sizes, int n_in,
                              void* d_out, int out_size, void* d_ws, size_t ws_size,
                              hipStream_t stream) {
  const int*   tok   = (const int*)  d_in[0];
  const float* h0    = (const float*)d_in[1];
  const float* c0    = (const float*)d_in[2];
  const float* enc   = (const float*)d_in[3];
  const float* emb   = (const float*)d_in[4];
  const float* W1    = (const float*)d_in[5];
  const float* W2    = (const float*)d_in[6];
  const float* v     = (const float*)d_in[7];
  const float* W_ih  = (const float*)d_in[8];
  const float* W_hh  = (const float*)d_in[9];
  const float* b_ih  = (const float*)d_in[10];
  const float* b_hh  = (const float*)d_in[11];
  const float* fc_W  = (const float*)d_in[12];
  const float* fc_b  = (const float*)d_in[13];

  float* out = (float*)d_out;
  float* out_pred = out;
  float* out_h    = out + (size_t)BATCH * VOCAB;
  float* out_c    = out_h + (size_t)BATCH * HID;
  float* out_attn = out_c + (size_t)BATCH * HID;

  // workspace layout (< 4 MB total)
  char* ws = (char*)d_ws;
  __bf16* W2b     = (__bf16*)(ws);                  // 2 MB
  __bf16* h0b     = (__bf16*)(ws + 2097152);        // 128 KB
  __bf16* embb    = (__bf16*)(ws + 2228224);        // 64 KB
  __bf16* ctxb    = (__bf16*)(ws + 2293760);        // 128 KB
  __bf16* h1b     = (__bf16*)(ws + 2424832);        // 128 KB
  float*  dh      = (float*) (ws + 2555904);        // 256 KB
  float*  scoresb = (float*) (ws + 2818048);        // 128 KB
  float*  gates   = (float*) (ws + 2949120);        // 1 MB

  // 1) conversions: W2 (L2-reused) and h0 -> bf16; zero scores accumulator
  k_cvt_bf16<<<dim3(1024), dim3(256), 0, stream>>>(W2, W2b, HID * HID / 4);
  k_cvt_bf16<<<dim3(64),   dim3(256), 0, stream>>>(h0, h0b, BATCH * HID / 4);
  k_zero<<<dim3(128), dim3(256), 0, stream>>>(scoresb, BATCH * SEQ);

  // 2) embedding gather (-> bf16)
  k_embed<<<dim3(BATCH), dim3(128), 0, stream>>>(tok, emb, embb);

  // 3) dh = h0 @ W1^T
  {
    GemmArgs g{};
    g.seg[0] = { h0b, W1, HID, HID, HID };
    g.nseg = 1;
    g.bias0 = nullptr; g.bias1 = nullptr;
    g.C = dh; g.ldc = HID; g.N = HID;
    k_gemm<<<dim3(HID / 256), dim3(256), 0, stream>>>(g);
  }

  // 4) scores += v . tanh(dh + enc @ W2^T)   (68.7 GFLOP, fused epilogue)
  k_scores<<<dim3(BATCH * SEQ / 64, 4), dim3(256), 0, stream>>>(enc, W2b, dh, v, scoresb);

  // 5) softmax + context (context kept as bf16 for downstream GEMM A-operands)
  k_softmax_ctx<<<dim3(BATCH), dim3(256), 0, stream>>>(scoresb, enc, out_attn, ctxb);

  // 6) gates = [embedded|context] @ W_ih^T + b_ih + h0 @ W_hh^T + b_hh
  {
    GemmArgs g{};
    g.seg[0] = { embb, W_ih,         EMBED, EMBED + HID, EMBED };
    g.seg[1] = { ctxb, W_ih + EMBED, HID,   EMBED + HID, HID   };
    g.seg[2] = { h0b,  W_hh,         HID,   HID,         HID   };
    g.nseg = 3;
    g.bias0 = b_ih; g.bias1 = b_hh;
    g.C = gates; g.ldc = 4 * HID; g.N = 4 * HID;
    k_gemm<<<dim3(4 * HID / 256), dim3(256), 0, stream>>>(g);
  }

  // 7) LSTM pointwise: c1, h1 (h1 also stored as bf16 for prediction GEMM)
  k_lstm<<<dim3(BATCH * HID / 256), dim3(256), 0, stream>>>(gates, c0, h1b, out_h, out_c);

  // 8) prediction = [h1|context] @ fc_W^T + fc_b   (fc_W streamed once: 262 MB)
  {
    GemmArgs g{};
    g.seg[0] = { h1b,  fc_W,       HID, 2 * HID, HID };
    g.seg[1] = { ctxb, fc_W + HID, HID, 2 * HID, HID };
    g.nseg = 2;
    g.bias0 = fc_b; g.bias1 = nullptr;
    g.C = out_pred; g.ldc = VOCAB; g.N = VOCAB;
    k_gemm<<<dim3(VOCAB / 256), dim3(256), 0, stream>>>(g);
  }
}
